// EquivariantBlock_30494267801866
// MI455X (gfx1250) — matrix operand order repaired and checked by reading, not compiled, not run on hardware
//
#include <hip/hip_runtime.h>
#include <stdint.h>

// ---------------------------------------------------------------------------
// EGNN EquivariantBlock for MI455X (gfx1250, wave32, WMMA).
//
// Fusion strategy (roofline): all [E,128] edge intermediates stay in LDS/regs;
// only agg (25.6MB, L2-resident) sees global atomics. Matrix ops use
// v_wmma_f32_16x16x32_bf16 (f32 accumulate). ~500 GFLOP total.
// All gathers/stages use 128-bit accesses (h rows are 256B contiguous).
//
// d_in index map — assumes depth-first insertion-order flattening of the
// setup_inputs() dict (params nested dict expanded in place).
// ---------------------------------------------------------------------------

#define N_NODES 50000
#define N_EDGES 1600000
#define HID     128
#define KP_EDGE 288   // 2*HID + 2 padded up to 9*32
#define KP_NODE 256   // 2*HID
#define EPB     32    // edges (or node rows) per block
#define TPB     64    // 2 waves of 32

typedef __attribute__((ext_vector_type(16))) __bf16 v16bf;
typedef __attribute__((ext_vector_type(8)))  __bf16 v8bf;
typedef __attribute__((ext_vector_type(8)))  float  v8f;

enum {
  IN_H = 0, IN_X, IN_EATTR, IN_NMASK, IN_EMASK,
  // gcl[0]
  IN_G0_EW1, IN_G0_EB1, IN_G0_EW2, IN_G0_EB2,
  IN_G0_NW1, IN_G0_NB1, IN_G0_NW2, IN_G0_NB2,
  IN_G0_ATTW, IN_G0_ATTB,
  // gcl[1]
  IN_G1_EW1, IN_G1_EB1, IN_G1_EW2, IN_G1_EB2,
  IN_G1_NW1, IN_G1_NB1, IN_G1_NW2, IN_G1_NB2,
  IN_G1_ATTW, IN_G1_ATTB,
  // equivariant update
  IN_EQ_W1, IN_EQ_B1, IN_EQ_W2, IN_EQ_B2, IN_EQ_W3,
  IN_EIDX
};

__device__ __forceinline__ float silu_f(float v) { return v * (1.0f / (1.0f + __expf(-v))); }
__device__ __forceinline__ float sigm_f(float v) { return 1.0f / (1.0f + __expf(-v)); }

__device__ __forceinline__ v8f vzero8() {
  v8f z;
#pragma unroll
  for (int i = 0; i < 8; ++i) z[i] = 0.0f;
  return z;
}

__device__ __forceinline__ v8f wmma_bf16(v16bf a, v16bf b, v8f c) {
  // D = A(16x32 bf16) x B(32x16 bf16) + C(16x16 f32)
  return __builtin_amdgcn_wmma_f32_16x16x32_bf16(false, a, false, b, (short)0, c, false, false);
}

// A fragment (16x32 bf16) from LDS row-major [row][k] per ISA 7.12.2:
// lane<16: elems0-7 K=k0+0..7,  elems8-15 K=k0+16..23 (M=lane)
// lane>=16: elems0-7 K=k0+8..15, elems8-15 K=k0+24..31 (M=lane-16)
__device__ __forceinline__ v16bf load_a_frag(const __bf16* base, int strideK,
                                             int m0, int k0, int lane) {
  const int m = lane & 15, hs = lane >> 4;
  const __bf16* p = base + (m0 + m) * strideK + k0;
  v8bf lo = *(const v8bf*)(p + hs * 8);
  v8bf hi = *(const v8bf*)(p + 16 + hs * 8);
  v16bf a;
#pragma unroll
  for (int i = 0; i < 8; ++i) { a[i] = lo[i]; a[i + 8] = hi[i]; }
  return a;
}

// B fragment (32x16 bf16): same K order as A: lanes 0-15 elems0-7 K=k0+0..7,
// elems8-15 K=k0+16..23; lanes 16-31 K=k0+8..15, k0+24..31. Weights pre-stored [nOut][Kpad].
__device__ __forceinline__ v16bf load_b_frag(const __bf16* wt, int Kpad,
                                             int n0, int k0, int lane) {
  const int n = lane & 15, hs = lane >> 4;
  const __bf16* p = wt + (size_t)(n0 + n) * Kpad + k0;
  v8bf lo = *(const v8bf*)(p + hs * 8);
  v8bf hi = *(const v8bf*)(p + 16 + hs * 8);
  v16bf b;
#pragma unroll
  for (int i = 0; i < 8; ++i) { b[i] = lo[i]; b[i + 8] = hi[i]; }
  return b;
}

// pack 4 f32 -> 4 bf16 (8 bytes)
union pk4 { __bf16 b[4]; uint2 u; };

// ---------------------------------------------------------------------------
// Fused edge kernel: gather -> MLP1 -> MLP2 -> (attention | eq scalar)
//                    -> atomic segment-sum.  IS_EQ picks the coord path.
// ---------------------------------------------------------------------------
template <bool IS_EQ>
__global__ __launch_bounds__(TPB)
void edge_kernel(const __bf16* __restrict__ hbf,
                 const float*  __restrict__ x,
                 const float*  __restrict__ edge_attr,
                 const float*  __restrict__ edge_mask,
                 const int*    __restrict__ edge_index,
                 const __bf16* __restrict__ w1t,   // [HID][KP_EDGE] bf16
                 const float*  __restrict__ b1,
                 const __bf16* __restrict__ w2t,   // [HID][HID] bf16
                 const float*  __restrict__ b2,
                 const float*  __restrict__ wv,    // att_w or eq_w3  [HID]
                 const float*  __restrict__ attb,  // att_b [1] (GCL only)
                 float*        __restrict__ agg)   // [N,HID] or xagg [N,3]
{
  __shared__ __align__(32) __bf16 sA[EPB * KP_EDGE];
  __shared__ __align__(32) __bf16 sHid[EPB * HID];
  __shared__ float sMij[EPB * HID];   // GCL only (dead weight in EQ instance)
  __shared__ float sScale[EPB];
  __shared__ float sCd[EPB * 4];
  __shared__ int   sRow[EPB];
  __shared__ int   sCol[EPB];
  __shared__ float sEm[EPB];

  const int tid  = threadIdx.x;
  const int lane = tid & 31;
  const int wave = tid >> 5;
  const int e0   = blockIdx.x * EPB;

  // --- phase 0: per-edge metadata + geometric features ----------------------
  if (tid < EPB) {
    const int ge = e0 + tid;
    const int r = edge_index[ge];
    const int c = edge_index[N_EDGES + ge];
    sRow[tid] = r; sCol[tid] = c;
    const float dx = x[r * 3 + 0] - x[c * 3 + 0];
    const float dy = x[r * 3 + 1] - x[c * 3 + 1];
    const float dz = x[r * 3 + 2] - x[c * 3 + 2];
    const float radial = dx * dx + dy * dy + dz * dz;
    const float inv = 1.0f / (sqrtf(radial + 1e-8f) + 1.0f);  // NORM_CONSTANT=1
    sCd[tid * 4 + 0] = dx * inv;
    sCd[tid * 4 + 1] = dy * inv;
    sCd[tid * 4 + 2] = dz * inv;
    sEm[tid] = edge_mask[ge];
    __bf16* row = &sA[tid * KP_EDGE];
    row[2 * HID + 0] = (__bf16)radial;
    row[2 * HID + 1] = (__bf16)edge_attr[ge];
#pragma unroll
    for (int k = 2 * HID + 2; k < KP_EDGE; ++k) row[k] = (__bf16)0.0f;
  }
  __syncthreads();

  // --- phase 1: gather h[row] || h[col] (bf16, 128-bit chunks) --------------
  {
    const uint4* h128 = (const uint4*)hbf;   // 16B = 8 bf16; row = 16 chunks
    for (int i = tid; i < EPB * (HID / 8); i += TPB) {
      const int e = i >> 4, j = i & 15;
      uint4* dst = (uint4*)&sA[e * KP_EDGE];   // 576B row stride, 16B aligned
      dst[j]      = h128[(size_t)sRow[e] * (HID / 8) + j];
      dst[16 + j] = h128[(size_t)sCol[e] * (HID / 8) + j];
    }
  }
  __syncthreads();

  const int m0 = wave * 16;
  const int nc = lane & 15;
  const int hs = lane >> 4;

  // --- phase 2: hid1 = silu(inp @ W1 + b1) ----------------------------------
  v8f acc[8];
#pragma unroll
  for (int t = 0; t < 8; ++t) acc[t] = vzero8();

#pragma unroll 1
  for (int kt = 0; kt < KP_EDGE / 32; ++kt) {
    const v16bf a = load_a_frag(sA, KP_EDGE, m0, kt * 32, lane);
#pragma unroll
    for (int nt = 0; nt < 8; ++nt) {
      const v16bf b = load_b_frag(w1t, KP_EDGE, nt * 16, kt * 32, lane);
      acc[nt] = wmma_bf16(a, b, acc[nt]);
    }
  }
#pragma unroll
  for (int nt = 0; nt < 8; ++nt) {
    const float bn = b1[nt * 16 + nc];
#pragma unroll
    for (int r = 0; r < 8; ++r) {
      const float v = silu_f(acc[nt][r] + bn);
      sHid[(m0 + r + 8 * hs) * HID + nt * 16 + nc] = (__bf16)v;
    }
  }
  __syncthreads();

  // --- phase 3: out = silu(hid1 @ W2 + b2), fused row-dot with wv -----------
  v8f acc2[8];
#pragma unroll
  for (int t = 0; t < 8; ++t) acc2[t] = vzero8();

#pragma unroll 1
  for (int kt = 0; kt < HID / 32; ++kt) {
    const v16bf a = load_a_frag(sHid, HID, m0, kt * 32, lane);
#pragma unroll
    for (int nt = 0; nt < 8; ++nt) {
      const v16bf b = load_b_frag(w2t, HID, nt * 16, kt * 32, lane);
      acc2[nt] = wmma_bf16(a, b, acc2[nt]);
    }
  }

  float dotp[8];
#pragma unroll
  for (int r = 0; r < 8; ++r) dotp[r] = 0.0f;

#pragma unroll
  for (int nt = 0; nt < 8; ++nt) {
    const int n = nt * 16 + nc;
    const float bn = b2[n];
    const float wn = wv[n];
#pragma unroll
    for (int r = 0; r < 8; ++r) {
      const float v = silu_f(acc2[nt][r] + bn);
      if constexpr (!IS_EQ) sMij[(m0 + r + 8 * hs) * HID + n] = v;
      dotp[r] += v * wn;
    }
  }
  // reduce dot over the 16 lanes of each half (wave32: masks stay in-half)
#pragma unroll
  for (int r = 0; r < 8; ++r) {
    float s = dotp[r];
    s += __shfl_xor(s, 1);
    s += __shfl_xor(s, 2);
    s += __shfl_xor(s, 4);
    s += __shfl_xor(s, 8);
    dotp[r] = s;
  }
  if (nc == 0) {
#pragma unroll
    for (int r = 0; r < 8; ++r) {
      const int e = m0 + r + 8 * hs;
      float s = dotp[r];
      if constexpr (!IS_EQ) s = sigm_f(s + attb[0]);   // attention
      sScale[e] = s * sEm[e];
    }
  }
  __syncthreads();

  // --- phase 4: atomic segment sum ------------------------------------------
  if constexpr (!IS_EQ) {
    for (int i = tid; i < EPB * HID; i += TPB) {
      const int e = i >> 7, n = i & (HID - 1);
      atomicAdd(&agg[(size_t)sRow[e] * HID + n], sMij[e * HID + n] * sScale[e]);
    }
  } else {
    for (int i = tid; i < EPB * 3; i += TPB) {
      const int e = i / 3, d = i - 3 * e;
      atomicAdd(&agg[(size_t)sRow[e] * 3 + d], sCd[e * 4 + d] * sScale[e]);
    }
  }
}

// ---------------------------------------------------------------------------
// Node kernel: h' = (h + silu([h, agg/100] @ n_w1 + b1) @ n_w2 + b2) * mask
// FINAL additionally applies the trailing h*node_mask of the reference.
// ---------------------------------------------------------------------------
template <bool FINAL>
__global__ __launch_bounds__(TPB)
void node_kernel(const float*  __restrict__ hin,
                 const float*  __restrict__ agg,
                 const float*  __restrict__ nmask,
                 const __bf16* __restrict__ w1t,   // [HID][KP_NODE]
                 const float*  __restrict__ b1,
                 const __bf16* __restrict__ w2t,   // [HID][HID]
                 const float*  __restrict__ b2,
                 float*        __restrict__ hout,
                 __bf16*       __restrict__ hbf_out)
{
  __shared__ __align__(32) __bf16 sA[EPB * KP_NODE];
  __shared__ __align__(32) __bf16 sHid[EPB * HID];

  const int tid  = threadIdx.x;
  const int lane = tid & 31;
  const int wave = tid >> 5;
  const int r0   = blockIdx.x * EPB;

  // stage [h, agg/100] as bf16 — float4 loads, 8B packed bf16 stores
  for (int i = tid; i < EPB * (HID / 4); i += TPB) {
    const int e = i >> 5, j = i & 31;          // j in units of 4 floats
    const int row = r0 + e;
    const int rc = row < N_NODES ? row : N_NODES - 1;
    const float4 hv = ((const float4*)hin)[(size_t)rc * (HID / 4) + j];
    const float4 av = ((const float4*)agg)[(size_t)rc * (HID / 4) + j];
    pk4 ph, pa;
    ph.b[0] = (__bf16)hv.x; ph.b[1] = (__bf16)hv.y;
    ph.b[2] = (__bf16)hv.z; ph.b[3] = (__bf16)hv.w;
    pa.b[0] = (__bf16)(av.x * 0.01f); pa.b[1] = (__bf16)(av.y * 0.01f);
    pa.b[2] = (__bf16)(av.z * 0.01f); pa.b[3] = (__bf16)(av.w * 0.01f);
    *(uint2*)&sA[e * KP_NODE + j * 4]       = ph.u;
    *(uint2*)&sA[e * KP_NODE + HID + j * 4] = pa.u;
  }
  __syncthreads();

  const int m0 = wave * 16;
  const int nc = lane & 15;
  const int hs = lane >> 4;

  v8f acc[8];
#pragma unroll
  for (int t = 0; t < 8; ++t) acc[t] = vzero8();
#pragma unroll 1
  for (int kt = 0; kt < KP_NODE / 32; ++kt) {
    const v16bf a = load_a_frag(sA, KP_NODE, m0, kt * 32, lane);
#pragma unroll
    for (int nt = 0; nt < 8; ++nt) {
      const v16bf b = load_b_frag(w1t, KP_NODE, nt * 16, kt * 32, lane);
      acc[nt] = wmma_bf16(a, b, acc[nt]);
    }
  }
#pragma unroll
  for (int nt = 0; nt < 8; ++nt) {
    const float bn = b1[nt * 16 + nc];
#pragma unroll
    for (int r = 0; r < 8; ++r) {
      const float v = silu_f(acc[nt][r] + bn);
      sHid[(m0 + r + 8 * hs) * HID + nt * 16 + nc] = (__bf16)v;
    }
  }
  __syncthreads();

  v8f acc2[8];
#pragma unroll
  for (int t = 0; t < 8; ++t) acc2[t] = vzero8();
#pragma unroll 1
  for (int kt = 0; kt < HID / 32; ++kt) {
    const v16bf a = load_a_frag(sHid, HID, m0, kt * 32, lane);
#pragma unroll
    for (int nt = 0; nt < 8; ++nt) {
      const v16bf b = load_b_frag(w2t, HID, nt * 16, kt * 32, lane);
      acc2[nt] = wmma_bf16(a, b, acc2[nt]);
    }
  }
#pragma unroll
  for (int nt = 0; nt < 8; ++nt) {
    const int n = nt * 16 + nc;
    const float bn = b2[n];
#pragma unroll
    for (int r = 0; r < 8; ++r) {
      const int m = m0 + r + 8 * hs;
      const int row = r0 + m;
      if (row < N_NODES) {
        const float nm = nmask[row];
        float hv = (hin[(size_t)row * HID + n] + acc2[nt][r] + bn) * nm;
        hbf_out[(size_t)row * HID + n] = (__bf16)hv;  // feeds next gathers
        if (FINAL) hv *= nm;                          // reference's final mask
        hout[(size_t)row * HID + n] = hv;
      }
    }
  }
}

// ---------------------------------------------------------------------------
// Small helper kernels
// ---------------------------------------------------------------------------
__global__ void cvt_w_kernel(const float* __restrict__ src,  // [K][HID]
                             __bf16* __restrict__ dst,       // [HID][Kpad]
                             int K, int Kpad) {
  const int i = blockIdx.x * blockDim.x + threadIdx.x;
  if (i >= HID * Kpad) return;
  const int n = i / Kpad, k = i - n * Kpad;
  dst[i] = (k < K) ? (__bf16)src[(size_t)k * HID + n] : (__bf16)0.0f;
}

__global__ void cvt_h_kernel(const float4* __restrict__ src,
                             uint2* __restrict__ dst, int total4) {
  const int i = blockIdx.x * blockDim.x + threadIdx.x;
  if (i >= total4) return;
  const float4 v = src[i];
  pk4 p;
  p.b[0] = (__bf16)v.x; p.b[1] = (__bf16)v.y;
  p.b[2] = (__bf16)v.z; p.b[3] = (__bf16)v.w;
  dst[i] = p.u;
}

__global__ void x_final_kernel(const float* __restrict__ x,
                               const float* __restrict__ xagg,
                               const float* __restrict__ nmask,
                               float* __restrict__ xout) {
  const int i = blockIdx.x * blockDim.x + threadIdx.x;
  if (i >= N_NODES * 3) return;
  xout[i] = (x[i] + xagg[i] * 0.01f) * nmask[i / 3];
}

// ---------------------------------------------------------------------------
extern "C" void kernel_launch(void* const* d_in, const int* in_sizes, int n_in,
                              void* d_out, int out_size, void* d_ws, size_t ws_size,
                              hipStream_t stream) {
  (void)in_sizes; (void)n_in; (void)out_size; (void)ws_size;

  const float* h0    = (const float*)d_in[IN_H];
  const float* x     = (const float*)d_in[IN_X];
  const float* eattr = (const float*)d_in[IN_EATTR];
  const float* nmask = (const float*)d_in[IN_NMASK];
  const float* emask = (const float*)d_in[IN_EMASK];
  const int*   eidx  = (const int*)d_in[IN_EIDX];

  // workspace carve-out (~66 MB needed)
  uint8_t* p = (uint8_t*)d_ws;
  auto take = [&](size_t n) -> void* {
    void* r = (void*)p;
    p += (n + 255) & ~(size_t)255;
    return r;
  };
  __bf16* hbf  = (__bf16*)take((size_t)N_NODES * HID * sizeof(__bf16));
  float*  h1   = (float*)take((size_t)N_NODES * HID * sizeof(float));
  float*  agg  = (float*)take((size_t)N_NODES * HID * sizeof(float));
  float*  xagg = (float*)take((size_t)N_NODES * 3 * sizeof(float));
  __bf16* ew1t[2], *ew2t[2], *nw1t[2], *nw2t[2];
  for (int l = 0; l < 2; ++l) {
    ew1t[l] = (__bf16*)take((size_t)HID * KP_EDGE * sizeof(__bf16));
    ew2t[l] = (__bf16*)take((size_t)HID * HID * sizeof(__bf16));
    nw1t[l] = (__bf16*)take((size_t)HID * KP_NODE * sizeof(__bf16));
    nw2t[l] = (__bf16*)take((size_t)HID * HID * sizeof(__bf16));
  }
  __bf16* eqw1t = (__bf16*)take((size_t)HID * KP_EDGE * sizeof(__bf16));
  __bf16* eqw2t = (__bf16*)take((size_t)HID * HID * sizeof(__bf16));

  float* h_out = (float*)d_out;                          // [N, HID]
  float* x_out = (float*)d_out + (size_t)N_NODES * HID;  // [N, 3]

  // --- weight conversion (bf16, transposed [N][Kpad]) -----------------------
  auto cvt = [&](int idx, __bf16* dst, int K, int Kpad) {
    const int tot = HID * Kpad;
    cvt_w_kernel<<<(tot + 255) / 256, 256, 0, stream>>>((const float*)d_in[idx],
                                                        dst, K, Kpad);
  };
  cvt(IN_G0_EW1, ew1t[0], 2 * HID + 2, KP_EDGE);
  cvt(IN_G0_EW2, ew2t[0], HID, HID);
  cvt(IN_G0_NW1, nw1t[0], KP_NODE, KP_NODE);
  cvt(IN_G0_NW2, nw2t[0], HID, HID);
  cvt(IN_G1_EW1, ew1t[1], 2 * HID + 2, KP_EDGE);
  cvt(IN_G1_EW2, ew2t[1], HID, HID);
  cvt(IN_G1_NW1, nw1t[1], KP_NODE, KP_NODE);
  cvt(IN_G1_NW2, nw2t[1], HID, HID);
  cvt(IN_EQ_W1, eqw1t, 2 * HID + 2, KP_EDGE);
  cvt(IN_EQ_W2, eqw2t, HID, HID);

  const int htot4 = N_NODES * HID / 4;
  cvt_h_kernel<<<(htot4 + 255) / 256, 256, 0, stream>>>((const float4*)h0,
                                                        (uint2*)hbf, htot4);

  const int egrid = N_EDGES / EPB;               // 50000, exact
  const int ngrid = (N_NODES + EPB - 1) / EPB;   // 1563

  // --- GCL layer 0 ----------------------------------------------------------
  hipMemsetAsync(agg, 0, (size_t)N_NODES * HID * sizeof(float), stream);
  edge_kernel<false><<<egrid, TPB, 0, stream>>>(
      hbf, x, eattr, emask, eidx,
      ew1t[0], (const float*)d_in[IN_G0_EB1],
      ew2t[0], (const float*)d_in[IN_G0_EB2],
      (const float*)d_in[IN_G0_ATTW], (const float*)d_in[IN_G0_ATTB], agg);
  node_kernel<false><<<ngrid, TPB, 0, stream>>>(
      h0, agg, nmask,
      nw1t[0], (const float*)d_in[IN_G0_NB1],
      nw2t[0], (const float*)d_in[IN_G0_NB2], h1, hbf);

  // --- GCL layer 1 ----------------------------------------------------------
  hipMemsetAsync(agg, 0, (size_t)N_NODES * HID * sizeof(float), stream);
  edge_kernel<false><<<egrid, TPB, 0, stream>>>(
      hbf, x, eattr, emask, eidx,
      ew1t[1], (const float*)d_in[IN_G1_EB1],
      ew2t[1], (const float*)d_in[IN_G1_EB2],
      (const float*)d_in[IN_G1_ATTW], (const float*)d_in[IN_G1_ATTB], agg);
  node_kernel<true><<<ngrid, TPB, 0, stream>>>(
      h1, agg, nmask,
      nw1t[1], (const float*)d_in[IN_G1_NB1],
      nw2t[1], (const float*)d_in[IN_G1_NB2], h_out, hbf);

  // --- Equivariant coordinate update ---------------------------------------
  hipMemsetAsync(xagg, 0, (size_t)N_NODES * 3 * sizeof(float), stream);
  edge_kernel<true><<<egrid, TPB, 0, stream>>>(
      hbf, x, eattr, emask, eidx,
      eqw1t, (const float*)d_in[IN_EQ_B1],
      eqw2t, (const float*)d_in[IN_EQ_B2],
      (const float*)d_in[IN_EQ_W3], nullptr, xagg);

  const int xtot = N_NODES * 3;
  x_final_kernel<<<(xtot + 255) / 256, 256, 0, stream>>>(x, xagg, nmask, x_out);
}
